// MultiHeadedAttention_71442486002120
// MI455X (gfx1250) — compile-verified, matrix operand order checked by reading, not run on hardware
//
#include <hip/hip_runtime.h>
#include <hip/hip_bf16.h>

typedef __attribute__((ext_vector_type(16))) _Float16 v16h;
typedef __attribute__((ext_vector_type(8)))  float    v8f;

#define WMMA_F16(A, B, C) \
  __builtin_amdgcn_wmma_f32_16x16x32_f16(false, (A), false, (B), (short)0, (C), false, false)

constexpr int Bb = 4, Ss = 2048, Ff = 512, Hh = 8, Dd = 64;
constexpr float MINV = -3.4028234663852886e38f;

union Frag16 { v16h h; uint4 u[2]; };

// ---------------------------------------------------------------------------
// LayerNorm over F=512 + cast to f16.  One block (256 threads) per row.
// ---------------------------------------------------------------------------
__global__ void __launch_bounds__(256)
ln_kernel(const float* __restrict__ x, const float* __restrict__ gamma,
          const float* __restrict__ beta, _Float16* __restrict__ xn)
{
  __shared__ float s0[8], s1[8];
  const int row = blockIdx.x;
  const int t   = threadIdx.x;
  const float* xr = x + (size_t)row * Ff;
  const float2 v2 = *reinterpret_cast<const float2*>(xr + t * 2);
  float s = v2.x + v2.y;
  float q = v2.x * v2.x + v2.y * v2.y;
#pragma unroll
  for (int off = 16; off >= 1; off >>= 1) {
    s += __shfl_xor(s, off, 32);
    q += __shfl_xor(q, off, 32);
  }
  const int wv = t >> 5;
  if ((t & 31) == 0) { s0[wv] = s; s1[wv] = q; }
  __syncthreads();
  if (t < 32) {
    float a = (t < 8) ? s0[t] : 0.f;
    float c = (t < 8) ? s1[t] : 0.f;
#pragma unroll
    for (int off = 4; off >= 1; off >>= 1) {
      a += __shfl_xor(a, off, 32);
      c += __shfl_xor(c, off, 32);
    }
    if (t == 0) { s0[0] = a; s1[0] = c; }
  }
  __syncthreads();
  const float mean = s0[0] * (1.0f / Ff);
  const float var  = s1[0] * (1.0f / Ff) - mean * mean;
  const float inv  = rsqrtf(var + 1e-5f);
  const int c = t * 2;
  const float y0 = (v2.x - mean) * inv * gamma[c]     + beta[c];
  const float y1 = (v2.y - mean) * inv * gamma[c + 1] + beta[c + 1];
  _Float16* o = xn + (size_t)row * Ff + c;
  o[0] = (_Float16)y0;
  o[1] = (_Float16)y1;
}

// ---------------------------------------------------------------------------
// fp32 -> f16 cast (weights)
// ---------------------------------------------------------------------------
__global__ void __launch_bounds__(256)
cvt_kernel(const float* __restrict__ src, _Float16* __restrict__ dst, int n)
{
  const int i = blockIdx.x * blockDim.x + threadIdx.x;
  if (i < n) dst[i] = (_Float16)src[i];
}

// ---------------------------------------------------------------------------
// Y[m,n] = (sum_k X[m,k]*W[n,k] + bias[n]) * scale
// X: MxK f16 row-major, W: NxK f16 row-major (einsum 'bsf,gf->bsg' == X @ W^T).
// Block = 128 thr = 4 waves stacked in M; each wave owns a 16x64 output tile
// (A operand reused across 4 WMMAs).  HEAD=true scatters to [B,H,S,D] f16.
// ---------------------------------------------------------------------------
template<bool HEAD, typename OutT>
__global__ void __launch_bounds__(128)
gemm16(const _Float16* __restrict__ X, const _Float16* __restrict__ W,
       const float* __restrict__ bias, OutT* __restrict__ out,
       int M, int N, int K, float scale)
{
  const int lane = threadIdx.x & 31;
  const int wv   = threadIdx.x >> 5;
  const int r    = lane & 15;
  const int hf   = lane >> 4;
  const int mBase = blockIdx.y * 64 + wv * 16;
  const int nBase = blockIdx.x * 64;

  v8f acc[4] = {};
  const _Float16* xrow = X + (size_t)(mBase + r) * K;
  const _Float16* wrow = W + (size_t)(nBase + r) * K;

  for (int k0 = 0; k0 < K; k0 += 32) {
    Frag16 a;
    a.u[0] = *reinterpret_cast<const uint4*>(xrow + k0 + hf * 8);
    a.u[1] = *reinterpret_cast<const uint4*>(xrow + k0 + 16 + hf * 8);
    __builtin_prefetch(xrow + k0 + 64, 0, 1);   // global_prefetch_b8 for next K step
#pragma unroll
    for (int t = 0; t < 4; ++t) {
      Frag16 bf;
      const _Float16* wr = wrow + (size_t)t * 16 * K;
      bf.u[0] = *reinterpret_cast<const uint4*>(wr + k0 + hf * 8);
      bf.u[1] = *reinterpret_cast<const uint4*>(wr + k0 + 16 + hf * 8);
      acc[t] = WMMA_F16(a.h, bf.h, acc[t]);
    }
  }

#pragma unroll
  for (int t = 0; t < 4; ++t) {
    const int n = nBase + t * 16 + r;
    const float bv = bias[n];
#pragma unroll
    for (int i = 0; i < 8; ++i) {
      const int m = mBase + i + 8 * hf;
      const float val = (acc[t][i] + bv) * scale;
      if (HEAD) {  // y.reshape(B,S,H,D).transpose(0,2,1,3)
        const int bi = m / Ss, si = m % Ss, hd = n >> 6, di = n & 63;
        out[(((size_t)bi * Hh + hd) * Ss + si) * Dd + di] = (OutT)val;
      } else {
        out[(size_t)m * N + n] = (OutT)val;
      }
    }
  }
}

// ---------------------------------------------------------------------------
// Flash attention per (b, h, 64-query block).  4 waves x 16 queries.
// S^T = K·Q^T so the P operand for ctx += P·V is already in A layout.
// K block (row-major, shared by all 4 waves) staged via async copy to LDS;
// V^T staged in LDS for the B operand of the second WMMA.
// ---------------------------------------------------------------------------
__global__ void __launch_bounds__(128)
attn_kernel(const _Float16* __restrict__ q, const _Float16* __restrict__ kmat,
            const _Float16* __restrict__ vmat, const int* __restrict__ mask,
            _Float16* __restrict__ ctxo)
{
  __shared__ _Float16 ldsK [64 * 64];   // K block, row-major [key][d]
  __shared__ _Float16 ldsVT[64 * 64];   // V block, transposed [d][key]

  const int lane = threadIdx.x & 31;
  const int wv   = threadIdx.x >> 5;
  const int r    = lane & 15;
  const int hf   = lane >> 4;
  const int qb   = blockIdx.x * 64;
  const int head = blockIdx.y;
  const int b    = blockIdx.z;
  const size_t bh = ((size_t)b * Hh + head) * Ss * Dd;

  // LDS byte offset of ldsK (addrspace(3) offset lives in the low 32 bits)
  const unsigned ldsKbase = (unsigned)(uintptr_t)(&ldsK[0]);

  // This wave's 16 Q rows (B operand of S^T; Q already pre-scaled by 1/sqrt(D))
  Frag16 qa0, qa1;
  {
    const _Float16* qrow = q + bh + (size_t)(qb + wv * 16 + r) * Dd;
    qa0.u[0] = *reinterpret_cast<const uint4*>(qrow + hf * 8);
    qa0.u[1] = *reinterpret_cast<const uint4*>(qrow + 16 + hf * 8);
    qa1.u[0] = *reinterpret_cast<const uint4*>(qrow + 32 + hf * 8);
    qa1.u[1] = *reinterpret_cast<const uint4*>(qrow + 48 + hf * 8);
  }

  v8f ct[4] = {};
  float mrun = MINV, lrun = 0.f;
  const int* mrow = mask + (size_t)b * Ss;

  for (int kb = 0; kb < Ss; kb += 64) {
    // ---- async copy of K block (8 KB) into LDS: fire-and-forget ----
    {
      const int tid = threadIdx.x;
#pragma unroll
      for (int j = 0; j < 4; ++j) {
        const int flat = tid + j * 128;         // 0..511 b128 chunks
        const int row  = flat >> 3;             // key row 0..63
        const int c8   = flat & 7;              // 16B chunk within row
        const _Float16* gp = kmat + bh + (size_t)(kb + row) * Dd + c8 * 8;
        const unsigned ldst = ldsKbase + (unsigned)(row * 128 + c8 * 16);
        asm volatile("global_load_async_to_lds_b128 %0, %1, off"
                     :: "v"(ldst), "v"(gp) : "memory");
      }
    }

    // ---- stage V^T[d, key] into LDS while the K copy is in flight ----
    {
      const int tid = threadIdx.x;
      const int row = tid >> 1;
      const int cs  = (tid & 1) * 32;
      const _Float16* vr = vmat + bh + (size_t)(kb + row) * Dd + cs;
      union { uint4 u[4]; _Float16 e[32]; } tmp;
      tmp.u[0] = *reinterpret_cast<const uint4*>(vr);
      tmp.u[1] = *reinterpret_cast<const uint4*>(vr + 8);
      tmp.u[2] = *reinterpret_cast<const uint4*>(vr + 16);
      tmp.u[3] = *reinterpret_cast<const uint4*>(vr + 24);
#pragma unroll
      for (int j = 0; j < 32; ++j) ldsVT[(cs + j) * 64 + row] = tmp.e[j];
    }

    asm volatile("s_wait_asynccnt 0x0" ::: "memory");
    __syncthreads();

    // ---- S^T tiles: 16 keys x 16 queries each, K-dim = D = 64 ----
    v8f st[4];
#pragma unroll
    for (int t = 0; t < 4; ++t) {
      const _Float16* krow = &ldsK[(t * 16 + r) * 64];
      Frag16 ka0, ka1;
      ka0.u[0] = *reinterpret_cast<const uint4*>(krow + hf * 8);
      ka0.u[1] = *reinterpret_cast<const uint4*>(krow + 16 + hf * 8);
      ka1.u[0] = *reinterpret_cast<const uint4*>(krow + 32 + hf * 8);
      ka1.u[1] = *reinterpret_cast<const uint4*>(krow + 48 + hf * 8);
      v8f z = {};
      z = WMMA_F16(ka0.h, qa0.h, z);
      z = WMMA_F16(ka1.h, qa1.h, z);
      st[t] = z;
    }

    // ---- mask: element (t, i) is key kb + t*16 + i + 8*hf ----
#pragma unroll
    for (int t = 0; t < 4; ++t)
#pragma unroll
      for (int i = 0; i < 8; ++i) {
        const int key = kb + t * 16 + i + 8 * hf;
        if (mrow[key] == 0) st[t][i] = MINV;
      }

    // ---- online softmax (each lane owns query index lane&15) ----
    float bm = MINV;
#pragma unroll
    for (int t = 0; t < 4; ++t)
#pragma unroll
      for (int i = 0; i < 8; ++i) bm = fmaxf(bm, st[t][i]);
    bm = fmaxf(bm, __shfl_xor(bm, 16, 32));
    const float mn    = fmaxf(mrun, bm);
    const float alpha = __expf(mrun - mn);

    float ps = 0.f;
    v16h pa0, pa1;   // P in A-operand layout: keys 0..31 / 32..63 of this block
#pragma unroll
    for (int i = 0; i < 8; ++i) {
      float p0 = __expf(st[0][i] - mn); ps += p0; pa0[i]     = (_Float16)p0;
      float p1 = __expf(st[1][i] - mn); ps += p1; pa0[i + 8] = (_Float16)p1;
      float p2 = __expf(st[2][i] - mn); ps += p2; pa1[i]     = (_Float16)p2;
      float p3 = __expf(st[3][i] - mn); ps += p3; pa1[i + 8] = (_Float16)p3;
    }
    lrun = lrun * alpha + (ps + __shfl_xor(ps, 16, 32));
    mrun = mn;

    // ---- rescale ctx accumulators (alpha broadcast per query row) ----
#pragma unroll
    for (int i = 0; i < 8; ++i) {
      const float av = __shfl(alpha, i + 8 * hf, 32);
#pragma unroll
      for (int t = 0; t < 4; ++t) ct[t][i] *= av;
    }

    // ---- ctx += P · V using V^T from LDS ----
#pragma unroll
    for (int t = 0; t < 4; ++t) {
      const _Float16* vt = &ldsVT[(t * 16 + r) * 64];
      Frag16 vb0, vb1;
      vb0.u[0] = *reinterpret_cast<const uint4*>(vt + hf * 8);
      vb0.u[1] = *reinterpret_cast<const uint4*>(vt + 16 + hf * 8);
      vb1.u[0] = *reinterpret_cast<const uint4*>(vt + 32 + hf * 8);
      vb1.u[1] = *reinterpret_cast<const uint4*>(vt + 48 + hf * 8);
      ct[t] = WMMA_F16(pa0, vb0.h, ct[t]);
      ct[t] = WMMA_F16(pa1, vb1.h, ct[t]);
    }
    __syncthreads();
  }

  // ---- normalize and store ctx as f16 in [B,S,F] for the final projection ----
  const float inv = 1.0f / lrun;
#pragma unroll
  for (int i = 0; i < 8; ++i) {
    const float sc = __shfl(inv, i + 8 * hf, 32);
    const int row = qb + wv * 16 + i + 8 * hf;
#pragma unroll
    for (int t = 0; t < 4; ++t) {
      ctxo[((size_t)b * Ss + row) * Ff + head * 64 + t * 16 + r] =
          (_Float16)(ct[t][i] * sc);
    }
  }
}

// ---------------------------------------------------------------------------
// Launcher
// ---------------------------------------------------------------------------
extern "C" void kernel_launch(void* const* d_in, const int* in_sizes, int n_in,
                              void* d_out, int out_size, void* d_ws, size_t ws_size,
                              hipStream_t stream)
{
  (void)in_sizes; (void)n_in; (void)out_size; (void)ws_size;

  const float* x     = (const float*)d_in[0];
  const int*   mask  = (const int*)d_in[1];
  const float* gamma = (const float*)d_in[2];
  const float* beta  = (const float*)d_in[3];
  const float* Wq    = (const float*)d_in[4];
  const float* bq    = (const float*)d_in[5];
  const float* Wk    = (const float*)d_in[6];
  const float* bk    = (const float*)d_in[7];
  const float* Wv    = (const float*)d_in[8];
  const float* bv    = (const float*)d_in[9];
  const float* Wo    = (const float*)d_in[10];
  const float* bo    = (const float*)d_in[11];
  float* out = (float*)d_out;

  // workspace layout (bytes)
  constexpr size_t SZ_XN  = (size_t)Bb * Ss * Ff * 2;        // 8 MiB
  constexpr size_t SZ_W   = (size_t)Ff * Ff * 2;             // 512 KiB
  constexpr size_t SZ_QKV = (size_t)Bb * Hh * Ss * Dd * 2;   // 8 MiB
  char* ws = (char*)d_ws;
  _Float16* xn   = (_Float16*)(ws);
  _Float16* wq16 = (_Float16*)(ws + SZ_XN);
  _Float16* wk16 = (_Float16*)(ws + SZ_XN + SZ_W);
  _Float16* wv16 = (_Float16*)(ws + SZ_XN + 2 * SZ_W);
  _Float16* wo16 = (_Float16*)(ws + SZ_XN + 3 * SZ_W);
  _Float16* qh   = (_Float16*)(ws + SZ_XN + 4 * SZ_W);
  _Float16* kh   = (_Float16*)(ws + SZ_XN + 4 * SZ_W + SZ_QKV);
  _Float16* vh   = (_Float16*)(ws + SZ_XN + 4 * SZ_W + 2 * SZ_QKV);
  _Float16* ctx  = (_Float16*)(ws + SZ_XN + 4 * SZ_W + 3 * SZ_QKV);

  const int M = Bb * Ss;        // 8192
  const int wN = Ff * Ff;       // 262144 elements per weight

  ln_kernel<<<M, 256, 0, stream>>>(x, gamma, beta, xn);

  cvt_kernel<<<(wN + 255) / 256, 256, 0, stream>>>(Wq, wq16, wN);
  cvt_kernel<<<(wN + 255) / 256, 256, 0, stream>>>(Wk, wk16, wN);
  cvt_kernel<<<(wN + 255) / 256, 256, 0, stream>>>(Wv, wv16, wN);
  cvt_kernel<<<(wN + 255) / 256, 256, 0, stream>>>(Wo, wo16, wN);

  const dim3 ggrid(Ff / 64, M / 64);   // (8, 128)
  gemm16<true, _Float16><<<ggrid, 128, 0, stream>>>(xn, wq16, bq, qh, M, Ff, Ff, 0.125f);
  gemm16<true, _Float16><<<ggrid, 128, 0, stream>>>(xn, wk16, bk, kh, M, Ff, Ff, 1.0f);
  gemm16<true, _Float16><<<ggrid, 128, 0, stream>>>(xn, wv16, bv, vh, M, Ff, Ff, 1.0f);

  attn_kernel<<<dim3(Ss / 64, Hh, Bb), 128, 0, stream>>>(qh, kh, vh, mask, ctx);

  gemm16<false, float><<<ggrid, 128, 0, stream>>>(ctx, wo16, bo, out, M, Ff, Ff, 1.0f);
}
